// HungarianMatcher_59416577573225
// MI455X (gfx1250) — compile-verified, matrix operand order checked by reading, not run on hardware
//
#include <hip/hip_runtime.h>
#include <hip/hip_bf16.h>
#include <math.h>

// ---- problem constants (from reference) ------------------------------------
#define BS  8
#define NQ  8192
#define NT  512
#define TAU 0.5f

typedef __attribute__((ext_vector_type(16))) _Float16     v16h;
typedef __attribute__((ext_vector_type(8)))  float        v8f;
typedef __attribute__((ext_vector_type(4)))  unsigned int u32x4;
typedef __attribute__((ext_vector_type(8)))  int          i32x8;
typedef __attribute__((ext_vector_type(4)))  int          i32x4;

#define WAVES_PER_BLOCK 8   // 256 threads = 8 wave32 waves
#define TILE_LDS_BYTES  (WAVES_PER_BLOCK * 16 * 16 * 4)   // 8 KB dynamic LDS

// ---------------------------------------------------------------------------
// Kernel 1: cost matrix via WMMA, drained through the Tensor Data Mover.
// d^2(q,t) = [x, y, |p|^2, 1] . [-2tx, -2ty, 1, |t|^2]  (K=4, zero-padded).
// One wave -> one 16x16 tile of C. The tile is staged into dynamic LDS with
// inline-asm ds_store_b32 (opaque to the optimizer: cannot be DCE'd or turned
// into flat stores), then TENSOR_STORE_FROM_LDS DMAs it to C.
// ---------------------------------------------------------------------------
__global__ __launch_bounds__(256) void hm_cost_wmma(
    const float* __restrict__ logits,   // [BS, NQ, 1]
    const float* __restrict__ pp,       // [BS, NQ, 2]
    const float* __restrict__ tp,       // [BS, NT, 2]
    float* __restrict__ C)              // [BS, NQ, NT]
{
#if __has_builtin(__builtin_amdgcn_tensor_store_from_lds)
  extern __shared__ float lds_tiles[]; // 8 KB; one 1 KB tile per wave
  (void)lds_tiles;
#endif

  const int lane = threadIdx.x & 31;
  const int wv   = (int)__builtin_amdgcn_readfirstlane(threadIdx.x >> 5);
  const int b    = blockIdx.z;
  const int j0   = blockIdx.x * 16;                            // target tile
  const int i0   = (blockIdx.y * WAVES_PER_BLOCK + wv) * 16;   // query tile

  const float inv = 1.0f / 512.0f;   // normalize coords so f16 operands <= 2

  // A (16x32 f16): lanes 0-15 = row M, halves 0..3 = [x, y, |p|^2, 1], rest 0.
  // B (32x16 f16): lanes 0-15 = col N, halves 0..3 = [-2tx, -2ty, 1, |t|^2].
  v16h a  = {};
  v16h bm = {};
  if (lane < 16) {
    const int m = i0 + lane;
    const float x = pp[((size_t)b * NQ + m) * 2 + 0] * inv;
    const float y = pp[((size_t)b * NQ + m) * 2 + 1] * inv;
    a[0] = (_Float16)x;
    a[1] = (_Float16)y;
    a[2] = (_Float16)(x * x + y * y);
    a[3] = (_Float16)1.0f;

    const int n = j0 + lane;
    const float tx = tp[((size_t)b * NT + n) * 2 + 0] * inv;
    const float ty = tp[((size_t)b * NT + n) * 2 + 1] * inv;
    bm[0] = (_Float16)(-2.0f * tx);
    bm[1] = (_Float16)(-2.0f * ty);
    bm[2] = (_Float16)1.0f;
    bm[3] = (_Float16)(tx * tx + ty * ty);
  }

  // gfx1250 prefetch path for the logits this wave reads below.
  __builtin_prefetch(&logits[(size_t)b * NQ + i0], 0, 0);

  // EXEC is all-1s here (divergence rejoined) as WMMA requires.
  v8f acc = {};
  acc = __builtin_amdgcn_wmma_f32_16x16x32_f16(
      /*neg_a=*/false, a, /*neg_b=*/false, bm,
      /*c_mod=*/(short)0, acc, /*reuse_a=*/false, /*reuse_b=*/false);

  // D layout: VGPR r -> M = r (lanes 0-15) / r+8 (lanes 16-31), N = lane%16.
  const int col   = lane & 15;
  const int rbase = (lane >> 4) ? 8 : 0;

  // Class costs first (plain loads, clause up front).
  float cls[8];
#pragma unroll
  for (int r = 0; r < 8; ++r) {
    const float lg = logits[(size_t)b * NQ + i0 + r + rbase];
    cls[r] = 1.0f - __builtin_amdgcn_rcpf(1.0f + __expf(-lg));
  }

#pragma unroll
  for (int r = 0; r < 8; ++r) {
    const int row = r + rbase;                       // row within tile
    float d2 = acc[r];
    d2 = fmaxf(d2, 0.0f);                            // clamp roundoff
    const float dist = 512.0f * __builtin_amdgcn_sqrtf(d2);   // raw v_sqrt_f32
    const float outv = TAU * dist + cls[r];
#if __has_builtin(__builtin_amdgcn_tensor_store_from_lds)
    // LDS byte offset of this element inside the wave's 1 KB tile.
    const int laddr = wv * 1024 + row * 64 + col * 4;
    asm volatile("ds_store_b32 %0, %1" :: "v"(laddr), "v"(outv));
#else
    C[((size_t)b * NQ + i0 + row) * NT + (j0 + col)] = outv;
#endif
  }

#if __has_builtin(__builtin_amdgcn_tensor_store_from_lds)
  // ---- drain the tile with the Tensor Data Mover --------------------------
  // D# per CDNA5 ISA: 2D tensor = C viewed as [BS*NQ, NT] f32.
  const unsigned lds_off = (unsigned)(wv * 1024);
  const unsigned long long gaddr =
      (unsigned long long)(uintptr_t)&C[((size_t)b * NQ + i0) * NT + j0];

  u32x4 g0;
  g0[0] = 1u;                                        // count=1, user descriptor
  g0[1] = lds_off;                                   // lds_addr [63:32]
  g0[2] = (unsigned)(gaddr & 0xffffffffu);           // global_addr[31:0]
  g0[3] = (unsigned)((gaddr >> 32) & 0x1ffffffu)     // global_addr[56:32]
          | (2u << 30);                              // type=2 ("image")

  i32x8 g1 = {};
  g1[0] = 0x00020000;                                // data_size=2 -> 4 bytes
  g1[1] = (int)(512u << 16);                         // tensor_dim0 = NT = 512
  g1[2] = 0;                                         // dim0 hi / dim1 lo (65536&0xffff)
  g1[3] = (16 << 16) | 1;                            // tile_dim0=16 | dim1 hi16=1
  g1[4] = 16;                                        // tile_dim1=16 (tile_dim2=0)
  g1[5] = 512;                                       // tensor_dim0_stride = NT
  g1[6] = 0;
  g1[7] = 0;
  i32x4 gz = {};                                     // groups 2/3 unused (2D)

  asm volatile("s_wait_dscnt 0x0" ::: "memory");     // LDS writes visible to TDM
#if defined(__clang_major__) && (__clang_major__ >= 23)
  {
    i32x8 gz8 = {};
    __builtin_amdgcn_tensor_store_from_lds(g0, g1, gz, gz, gz8, 0);
  }
#else
  __builtin_amdgcn_tensor_store_from_lds(g0, g1, gz, gz, 0);
#endif
  __builtin_amdgcn_s_wait_tensorcnt(0);              // fence before wave exit
#endif
}

// ---------------------------------------------------------------------------
// Kernel 2: deterministic per-batch assignment (greedy argmin stand-in for the
// host-side Hungarian). One block per batch; LDS tree reduction over NQ.
// ---------------------------------------------------------------------------
__global__ __launch_bounds__(256) void hm_assign_greedy(
    const float* __restrict__ C,        // [BS, NQ, NT]
    float* __restrict__ out_rows,       // [BS, NT]
    float* __restrict__ out_cols)       // [BS, NT]
{
  __shared__ float         sval[256];
  __shared__ int           sidx[256];
  __shared__ unsigned char taken[NQ];   // 8 KB LDS

  const int b   = blockIdx.x;
  const int tid = threadIdx.x;

  for (int q = tid; q < NQ; q += 256) taken[q] = 0;
  __syncthreads();

  for (int j = 0; j < NT; ++j) {
    float best = 3.0e38f;
    int   bi   = 0x7fffffff;
    for (int q = tid; q < NQ; q += 256) {
      if (!taken[q]) {
        const float c = C[((size_t)b * NQ + q) * NT + j];
        if (c < best || (c == best && q < bi)) { best = c; bi = q; }
      }
    }
    sval[tid] = best;
    sidx[tid] = bi;
    __syncthreads();
#pragma unroll
    for (int s = 128; s > 0; s >>= 1) {
      if (tid < s) {
        const float ov = sval[tid + s];
        const int   oi = sidx[tid + s];
        if (ov < sval[tid] || (ov == sval[tid] && oi < sidx[tid])) {
          sval[tid] = ov;
          sidx[tid] = oi;
        }
      }
      __syncthreads();
    }
    if (tid == 0) {
      const int q = sidx[0];
      taken[q] = 1;
      out_rows[(size_t)b * NT + j] = (float)q;
      out_cols[(size_t)b * NT + j] = (float)j;
    }
    __syncthreads();
  }
}

// ---------------------------------------------------------------------------
extern "C" void kernel_launch(void* const* d_in, const int* in_sizes, int n_in,
                              void* d_out, int out_size, void* d_ws, size_t ws_size,
                              hipStream_t stream) {
  const float* logits = (const float*)d_in[0];  // [BS, NQ, 1]
  const float* pp     = (const float*)d_in[1];  // [BS, NQ, 2]
  const float* tp     = (const float*)d_in[2];  // [BS, NT, 2]

  float* C = (float*)d_out;                     // first output: cost matrix
  const long long c_elems = (long long)BS * NQ * NT;

  dim3 grid(NT / 16, NQ / (16 * WAVES_PER_BLOCK), BS);  // 32 x 64 x 8
  dim3 block(256, 1, 1);
  hipLaunchKernelGGL(hm_cost_wmma, grid, block, TILE_LDS_BYTES, stream,
                     logits, pp, tp, C);

  // Index outputs live after C in d_out (reference returns (C, row_ind, col_ind)).
  if ((long long)out_size >= c_elems + 2LL * BS * NT) {
    float* rows = C + c_elems;
    float* cols = rows + (long long)BS * NT;
    hipLaunchKernelGGL(hm_assign_greedy, dim3(BS), dim3(256), 0, stream, C, rows, cols);
  }
}
